// AgentPolicyNetV2_79559974191123
// MI455X (gfx1250) — matrix operand order repaired and checked by reading, not compiled, not run on hardware
//
#include <hip/hip_runtime.h>
#include <hip/hip_bf16.h>
#include <math.h>

// Problem constants (from reference)
#define B_    2048
#define S_    200
#define H_    256
#define E_    64
#define NH_   4
#define I_    16
#define L_    2
#define PREF_ 32
#define REMB_ 128
#define TEMB_ 32
#define C_    8
#define IN_   448          // REMB*2 + TEMB + 2*E + PREF
#define LN_EPS 1e-5f

typedef __attribute__((ext_vector_type(16))) __bf16 v16bf;
typedef __attribute__((ext_vector_type(8)))  __bf16 v8bf;
typedef __attribute__((ext_vector_type(8)))  float  v8f;

__device__ __forceinline__ __bf16 f2bf(float f) {
  unsigned u = __builtin_bit_cast(unsigned, f);
  unsigned r = u + 0x7FFFu + ((u >> 16) & 1u);       // round-to-nearest-even
  unsigned short s = (unsigned short)(r >> 16);
  return __builtin_bit_cast(__bf16, s);
}
__device__ __forceinline__ unsigned short bf_bits(float f) {
  unsigned u = __builtin_bit_cast(unsigned, f);
  unsigned r = u + 0x7FFFu + ((u >> 16) & 1u);
  return (unsigned short)(r >> 16);
}

// A fragment: 16x32 bf16 tile of row-major bf16 A (lda in elements).
// ISA layout: lanes 0-15 = rows; elems 0-7 -> K = k0+8h.., elems 8-15 -> K = k0+16+8h..
__device__ __forceinline__ v16bf load_a_bf(const __bf16* __restrict__ A, int lda,
                                           int row, int k0, int lane) {
  const int hf = lane >> 4;
  const __bf16* p = A + (long)row * lda + k0 + 8 * hf;
  v8bf lo = *(const v8bf*)p;          // 16B aligned contiguous run
  v8bf hi = *(const v8bf*)(p + 16);
  return __builtin_shufflevector(lo, hi, 0,1,2,3,4,5,6,7,8,9,10,11,12,13,14,15);
}

// B fragment: 32x16 where B[k][n] = W[n][k], W row-major (N,K) bf16.
// ISA layout: elems 0-7 -> K = k0+8*kg.., elems 8-15 -> K = k0+16+8*kg.. (matches A).
__device__ __forceinline__ v16bf load_b_bf(const __bf16* __restrict__ W, int ldw,
                                           int col, int k0, int lane) {
  const int kg = lane >> 4;
  const __bf16* p = W + (long)col * ldw + k0 + 8 * kg;
  v8bf lo = *(const v8bf*)p;
  v8bf hi = *(const v8bf*)(p + 16);
  return __builtin_shufflevector(lo, hi, 0,1,2,3,4,5,6,7,8,9,10,11,12,13,14,15);
}

// C-tile stores (avoid accumulator pointer arrays -> no register shuffling)
__device__ __forceinline__ void store_relu_bf(const v8f& c, __bf16* __restrict__ Y,
                                              const float* __restrict__ bias,
                                              int m, int n) {
  const float bn = bias[n];
#pragma unroll
  for (int r = 0; r < 8; ++r) {
    const float v = c[r] + bn;
    Y[(long)(m + r) * H_ + n] = f2bf(v > 0.f ? v : 0.f);
  }
}
__device__ __forceinline__ void store_gates(const v8f& c, float* __restrict__ G,
                                            const float* __restrict__ bih,
                                            const float* __restrict__ bhh,
                                            int m, int n) {
  const float bn = bih[n] + bhh[n];
#pragma unroll
  for (int r = 0; r < 8; ++r)
    G[(long)(m + r) * (4 * H_) + n] = c[r] + bn;
}

// ---------------- f32 -> bf16 bulk conversion (packed 32-bit stores) --------
__global__ void k_tobf(const float* __restrict__ in, unsigned* __restrict__ out2, int n2) {
  const int i = blockIdx.x * blockDim.x + threadIdx.x;
  if (i < n2) {
    const float a = in[2 * i], b = in[2 * i + 1];
    out2[i] = (unsigned)bf_bits(a) | ((unsigned)bf_bits(b) << 16);
  }
}

// ---------------- Stage 0: small attention + LN + embedding concat ----------
__global__ void k_prep(const int* __restrict__ loc, const int* __restrict__ tim,
                       const int* __restrict__ des, const float* __restrict__ inter,
                       const float* __restrict__ prefer,
                       const float* __restrict__ remb, const float* __restrict__ temb,
                       const float* __restrict__ Wq, const float* __restrict__ bq,
                       const float* __restrict__ Wk, const float* __restrict__ bk,
                       const float* __restrict__ Wv, const float* __restrict__ bv,
                       const float* __restrict__ Wo, const float* __restrict__ bo,
                       const float* __restrict__ ln_g, const float* __restrict__ ln_b,
                       __bf16* __restrict__ lstm_in) {
  __shared__ float s_in[I_ * E_];
  __shared__ float s_Q[E_];
  __shared__ float s_K[I_ * E_];
  __shared__ float s_V[I_ * E_];
  __shared__ float s_att[NH_ * I_];
  __shared__ float s_ctx[E_];
  __shared__ float s_cat[2 * E_];
  __shared__ float s_red[E_];
  const int b = blockIdx.x, tid = threadIdx.x;   // 64 threads
  for (int i = tid; i < I_ * E_; i += E_) s_in[i] = inter[(long)b * I_ * E_ + i];
  __syncthreads();
  { // Q row (from token 0)
    float q = bq[tid];
    const float* w = Wq + tid * E_;
    for (int k = 0; k < E_; ++k) q += w[k] * s_in[k];
    s_Q[tid] = q;
  }
  // K,V: thread computes output feature `tid` for all 16 tokens
  for (int i = 0; i < I_; ++i) {
    float kk = bk[tid], vv = bv[tid];
    const float* wk = Wk + tid * E_;
    const float* wv = Wv + tid * E_;
    const float* xi = s_in + i * E_;
    for (int k = 0; k < E_; ++k) { kk += wk[k] * xi[k]; vv += wv[k] * xi[k]; }
    s_K[i * E_ + tid] = kk; s_V[i * E_ + tid] = vv;
  }
  __syncthreads();
  { // logits (NH=4 heads, dh=16)
    const int h = tid >> 4, i = tid & 15;
    float lg = 0.f;
    for (int d = 0; d < 16; ++d) lg += s_Q[h * 16 + d] * s_K[i * E_ + h * 16 + d];
    s_att[h * I_ + i] = lg * 0.25f;                 // 1/sqrt(16)
  }
  __syncthreads();
  if (tid < NH_) {                                  // per-head softmax over 16
    float mx = -1e30f;
    for (int i = 0; i < I_; ++i) mx = fmaxf(mx, s_att[tid * I_ + i]);
    float sum = 0.f;
    for (int i = 0; i < I_; ++i) { float e = __expf(s_att[tid * I_ + i] - mx); s_att[tid * I_ + i] = e; sum += e; }
    float inv = 1.f / sum;
    for (int i = 0; i < I_; ++i) s_att[tid * I_ + i] *= inv;
  }
  __syncthreads();
  { // ctx
    const int h = tid >> 4, d = tid & 15;
    float acc = 0.f;
    for (int i = 0; i < I_; ++i) acc += s_att[h * I_ + i] * s_V[i * E_ + h * 16 + d];
    s_ctx[tid] = acc;
  }
  __syncthreads();
  { // attn_out + concat
    float ao = bo[tid];
    const float* w = Wo + tid * E_;
    for (int k = 0; k < E_; ++k) ao += w[k] * s_ctx[k];
    s_cat[tid] = s_in[tid];
    s_cat[E_ + tid] = ao;
  }
  __syncthreads();
  // LayerNorm over 128
  s_red[tid] = s_cat[tid] + s_cat[tid + E_];
  __syncthreads();
  for (int off = 32; off > 0; off >>= 1) { if (tid < off) s_red[tid] += s_red[tid + off]; __syncthreads(); }
  const float mu = s_red[0] * (1.f / 128.f);
  __syncthreads();
  const float d0 = s_cat[tid] - mu, d1 = s_cat[tid + E_] - mu;
  s_red[tid] = d0 * d0 + d1 * d1;
  __syncthreads();
  for (int off = 32; off > 0; off >>= 1) { if (tid < off) s_red[tid] += s_red[tid + off]; __syncthreads(); }
  const float rstd = rsqrtf(s_red[0] * (1.f / 128.f) + LN_EPS);
  __syncthreads();
  // Assemble bf16 lstm_in row: [road_e(128), time_e(32), prefer(32), des_e(128), info(128)]
  __bf16* dst = lstm_in + (long)b * IN_;
  const int li = loc[b], ti = tim[b], di = des[b];
  for (int j = tid; j < REMB_; j += E_) dst[j] = f2bf(remb[(long)li * REMB_ + j]);
  if (tid < TEMB_) dst[REMB_ + tid] = f2bf(temb[(long)ti * TEMB_ + tid]);
  if (tid < PREF_) dst[REMB_ + TEMB_ + tid] = f2bf(prefer[(long)b * PREF_ + tid]);
  for (int j = tid; j < REMB_; j += E_) dst[192 + j] = f2bf(remb[(long)di * REMB_ + j]);
  dst[320 + tid]      = f2bf(d0 * rstd * ln_g[tid]      + ln_b[tid]);
  dst[320 + E_ + tid] = f2bf(d1 * rstd * ln_g[E_ + tid] + ln_b[E_ + tid]);
}

// ---------------- Stage 1: fc_in (WMMA bf16, 16x64 strip/wave, bias+ReLU) ---
__global__ void k_fcin(const __bf16* __restrict__ X, const __bf16* __restrict__ W,
                       const float* __restrict__ bias, __bf16* __restrict__ Y) {
  const int lane = threadIdx.x;
  const int tn = blockIdx.x * 64, tm = blockIdx.y * 16;
  const int row = tm + (lane & 15);
  const int nc  = lane & 15;
  v8f c0 = {}, c1 = {}, c2 = {}, c3 = {};
  for (int k0 = 0; k0 < IN_; k0 += 32) {
    v16bf a = load_a_bf(X, IN_, row, k0, lane);
    __builtin_prefetch(X + (long)row * IN_ + k0 + 32, 0, 1);
    v16bf b0 = load_b_bf(W, IN_, tn +      nc, k0, lane);
    v16bf b1 = load_b_bf(W, IN_, tn + 16 + nc, k0, lane);
    v16bf b2 = load_b_bf(W, IN_, tn + 32 + nc, k0, lane);
    v16bf b3 = load_b_bf(W, IN_, tn + 48 + nc, k0, lane);
    c0 = __builtin_amdgcn_wmma_f32_16x16x32_bf16(false, a, false, b0, (short)0, c0, false, false);
    c1 = __builtin_amdgcn_wmma_f32_16x16x32_bf16(false, a, false, b1, (short)0, c1, false, false);
    c2 = __builtin_amdgcn_wmma_f32_16x16x32_bf16(false, a, false, b2, (short)0, c2, false, false);
    c3 = __builtin_amdgcn_wmma_f32_16x16x32_bf16(false, a, false, b3, (short)0, c3, false, false);
  }
  const int m = tm + 8 * (lane >> 4);
  store_relu_bf(c0, Y, bias, m, tn +      nc);
  store_relu_bf(c1, Y, bias, m, tn + 16 + nc);
  store_relu_bf(c2, Y, bias, m, tn + 32 + nc);
  store_relu_bf(c3, Y, bias, m, tn + 48 + nc);
}

// ---------------- Stage 2a: LSTM gates (WMMA bf16, K=512 split) -------------
__global__ void k_gates(const __bf16* __restrict__ X, const __bf16* __restrict__ Hc,
                        const __bf16* __restrict__ Wih, const __bf16* __restrict__ Whh,
                        const float* __restrict__ bih, const float* __restrict__ bhh,
                        float* __restrict__ G) {
  const int lane = threadIdx.x;
  const int tn = blockIdx.x * 64, tm = blockIdx.y * 16;
  const int row = tm + (lane & 15);
  const int nc  = lane & 15;
  v8f c0 = {}, c1 = {}, c2 = {}, c3 = {};
  for (int k0 = 0; k0 < H_; k0 += 32) {        // x @ Wih^T
    v16bf a = load_a_bf(X, H_, row, k0, lane);
    __builtin_prefetch(X + (long)row * H_ + k0 + 32, 0, 1);
    v16bf b0 = load_b_bf(Wih, H_, tn +      nc, k0, lane);
    v16bf b1 = load_b_bf(Wih, H_, tn + 16 + nc, k0, lane);
    v16bf b2 = load_b_bf(Wih, H_, tn + 32 + nc, k0, lane);
    v16bf b3 = load_b_bf(Wih, H_, tn + 48 + nc, k0, lane);
    c0 = __builtin_amdgcn_wmma_f32_16x16x32_bf16(false, a, false, b0, (short)0, c0, false, false);
    c1 = __builtin_amdgcn_wmma_f32_16x16x32_bf16(false, a, false, b1, (short)0, c1, false, false);
    c2 = __builtin_amdgcn_wmma_f32_16x16x32_bf16(false, a, false, b2, (short)0, c2, false, false);
    c3 = __builtin_amdgcn_wmma_f32_16x16x32_bf16(false, a, false, b3, (short)0, c3, false, false);
  }
  for (int k0 = 0; k0 < H_; k0 += 32) {        // + h @ Whh^T
    v16bf a = load_a_bf(Hc, H_, row, k0, lane);
    __builtin_prefetch(Hc + (long)row * H_ + k0 + 32, 0, 1);
    v16bf b0 = load_b_bf(Whh, H_, tn +      nc, k0, lane);
    v16bf b1 = load_b_bf(Whh, H_, tn + 16 + nc, k0, lane);
    v16bf b2 = load_b_bf(Whh, H_, tn + 32 + nc, k0, lane);
    v16bf b3 = load_b_bf(Whh, H_, tn + 48 + nc, k0, lane);
    c0 = __builtin_amdgcn_wmma_f32_16x16x32_bf16(false, a, false, b0, (short)0, c0, false, false);
    c1 = __builtin_amdgcn_wmma_f32_16x16x32_bf16(false, a, false, b1, (short)0, c1, false, false);
    c2 = __builtin_amdgcn_wmma_f32_16x16x32_bf16(false, a, false, b2, (short)0, c2, false, false);
    c3 = __builtin_amdgcn_wmma_f32_16x16x32_bf16(false, a, false, b3, (short)0, c3, false, false);
  }
  const int m = tm + 8 * (lane >> 4);
  store_gates(c0, G, bih, bhh, m, tn +      nc);
  store_gates(c1, G, bih, bhh, m, tn + 16 + nc);
  store_gates(c2, G, bih, bhh, m, tn + 32 + nc);
  store_gates(c3, G, bih, bhh, m, tn + 48 + nc);
}

// ---------------- Stage 2b: LSTM cell elementwise ---------------------------
__global__ void k_cell(const float* __restrict__ G, const float* __restrict__ c_in,
                       float* __restrict__ h_out, float* __restrict__ c_out,
                       __bf16* __restrict__ x_bf, float* __restrict__ x_f32) {
  const int idx = blockIdx.x * blockDim.x + threadIdx.x;
  if (idx >= B_ * H_) return;
  const int b = idx / H_, j = idx % H_;
  const float* g = G + (long)b * 4 * H_;
  const float ig = g[j], fg = g[H_ + j], gg = g[2 * H_ + j], og = g[3 * H_ + j];
  const float si = 1.f / (1.f + __expf(-ig));
  const float sf = 1.f / (1.f + __expf(-fg));
  const float so = 1.f / (1.f + __expf(-og));
  const float cn = sf * c_in[idx] + si * tanhf(gg);
  const float hn = so * tanhf(cn);
  c_out[idx] = cn; h_out[idx] = hn;
  x_bf[idx] = f2bf(hn); x_f32[idx] = hn;
}

// ---------------- Stage 3: history attention (async HBM->LDS staging) -------
__global__ void k_hist(const float* __restrict__ hist, const float* __restrict__ lstm_out,
                       float* __restrict__ out_hist, float* __restrict__ moving) {
  extern __shared__ float smem[];
  float* sh = smem;                    // (S_+1)*H_ staged history row
  float* so = smem + (S_ + 1) * H_;    // H_   lstm_out
  float* sr = so + H_;                 // H_   reduction scratch
  float* sw = sr + H_;                 // H_   softmax weights (201 used)
  const int b = blockIdx.x, tid = threadIdx.x;   // 256 threads
  const float* hrow = hist + (long)b * S_ * H_;
  float* orow = out_hist + (long)b * (S_ + 1) * H_;
  // Async bulk fill HBM -> LDS, 16B per op, no VGPR staging (ASYNCcnt-tracked).
  // Low 32 bits of a generic LDS pointer are the LDS byte address (aperture rule).
  const unsigned lds0 = (unsigned)(size_t)sh;
  for (int i = tid; i < S_ * H_ / 4; i += 256) {
    const float* g = hrow + 4 * i;
    const unsigned loff = lds0 + 16u * i;
    asm volatile("global_load_async_to_lds_b128 %0, %1, off"
                 :: "v"(loff), "v"(g) : "memory");
  }
  const float lo = lstm_out[(long)b * H_ + tid];
  so[tid] = lo;
  asm volatile("s_wait_asynccnt 0x0" ::: "memory");
  __syncthreads();                      // all waves' async fills visible
  sh[S_ * H_ + tid] = lo;
  orow[S_ * H_ + tid] = lo;
  // Emit next_history_h rows 0..199 from LDS (b128 in, b128 out)
  {
    const float4* s4 = (const float4*)sh;
    float4* o4 = (float4*)orow;
    for (int i = tid; i < S_ * H_ / 4; i += 256) o4[i] = s4[i];
  }
  __syncthreads();
  // logits over 201 steps (thread tid handles step tid)
  float lg = -1e30f;
  if (tid <= S_) {
    float acc = 0.f;
    const float* hs = sh + tid * H_;
    for (int k = 0; k < H_; ++k) acc += hs[k] * so[k];
    lg = acc * (1.f / 16.f);                       // 1/sqrt(256)
  }
  sr[tid] = lg; __syncthreads();
  for (int off = 128; off > 0; off >>= 1) { if (tid < off) sr[tid] = fmaxf(sr[tid], sr[tid + off]); __syncthreads(); }
  const float mx = sr[0]; __syncthreads();
  float ex = 0.f;
  if (tid <= S_) { ex = __expf(lg - mx); sw[tid] = ex; }
  sr[tid] = ex; __syncthreads();
  for (int off = 128; off > 0; off >>= 1) { if (tid < off) sr[tid] += sr[tid + off]; __syncthreads(); }
  const float inv = 1.f / sr[0]; __syncthreads();
  float acc = 0.f;
  for (int s = 0; s <= S_; ++s) acc += sw[s] * sh[s * H_ + tid];
  moving[(long)b * H_ + tid] = acc * inv;
}

// ---------------- Stage 4: candidate gather head ----------------------------
__global__ void k_cand(const float* __restrict__ moving, const int* __restrict__ cmask,
                       const float* __restrict__ outW, const float* __restrict__ outb,
                       float* __restrict__ prob) {
  const int b = blockIdx.x;
  const int c = threadIdx.x >> 5, lane = threadIdx.x & 31;
  const int idx = cmask[b * C_ + c];
  const float* w = outW + (long)idx * H_;
  const float* m = moving + (long)b * H_;
  float acc = 0.f;
  for (int k = lane; k < H_; k += 32) acc += m[k] * w[k];
  for (int off = 16; off > 0; off >>= 1) acc += __shfl_down(acc, off, 32);
  if (lane == 0) prob[b * C_ + c] = acc + outb[idx];
}

// ---------------- Stage 5: out_info head ------------------------------------
__global__ void k_info(const float* __restrict__ moving, const float* __restrict__ W,
                       const float* __restrict__ bias, float* __restrict__ out) {
  const int b = blockIdx.x, e = threadIdx.x;     // 64 threads
  const float* m = moving + (long)b * H_;
  const float* w = W + (long)e * H_;
  float acc = bias[e];
  for (int k = 0; k < H_; ++k) acc += m[k] * w[k];
  out[(long)b * E_ + e] = acc;
}

extern "C" void kernel_launch(void* const* d_in, const int* in_sizes, int n_in,
                              void* d_out, int out_size, void* d_ws, size_t ws_size,
                              hipStream_t stream) {
  (void)in_sizes; (void)n_in; (void)out_size; (void)ws_size;
  const int*   loc    = (const int*)d_in[0];
  const int*   tim    = (const int*)d_in[1];
  const int*   des    = (const int*)d_in[2];
  const float* inter  = (const float*)d_in[3];
  const float* prefer = (const float*)d_in[4];
  const int*   cmask  = (const int*)d_in[5];
  const float* hist   = (const float*)d_in[6];
  const float* cur_h  = (const float*)d_in[7];
  const float* cur_c  = (const float*)d_in[8];
  const float* remb   = (const float*)d_in[9];
  const float* temb   = (const float*)d_in[10];
  const float* Wq = (const float*)d_in[11]; const float* bq = (const float*)d_in[12];
  const float* Wk = (const float*)d_in[13]; const float* bk = (const float*)d_in[14];
  const float* Wv = (const float*)d_in[15]; const float* bv = (const float*)d_in[16];
  const float* Wo = (const float*)d_in[17]; const float* bo = (const float*)d_in[18];
  const float* ln_g = (const float*)d_in[19]; const float* ln_b = (const float*)d_in[20];
  const float* fcW = (const float*)d_in[21]; const float* fcb = (const float*)d_in[22];
  const float* Wih = (const float*)d_in[23]; const float* Whh = (const float*)d_in[24];
  const float* bih = (const float*)d_in[25]; const float* bhh = (const float*)d_in[26];
  const float* outW = (const float*)d_in[27]; const float* outb = (const float*)d_in[28];
  const float* infW = (const float*)d_in[29]; const float* infb = (const float*)d_in[30];

  // Output layout (flat concatenation, reference return order)
  float* out    = (float*)d_out;
  float* o_cand = out;                         // B*C          = 16384
  float* o_info = out + 16384;                 // B*E          = 131072
  float* o_hist = out + 147456;                // B*201*H      = 105381888
  float* o_nh   = out + 105529344;             // L*B*H        = 1048576
  float* o_nc   = out + 106577920;             // L*B*H        = 1048576
  float* o_ms   = out + 107626496;             // B*H          = 524288

  // Workspace layout (byte offsets, all 16B aligned)
  char* ws = (char*)d_ws;
  float*  ws_g     = (float*)(ws);                               // B*4H f32 gates  (8 MB)
  __bf16* ws_in_bf = (__bf16*)(ws + 8388608);                    // B*448 bf16
  __bf16* ws_x_bf  = (__bf16*)(ws + 8388608 + 1835008);          // B*256 bf16
  float*  ws_xf    = (float*)(ws + 8388608 + 1835008 + 1048576); // B*256 f32 lstm_out
  __bf16* ws_h_bf  = (__bf16*)(ws + 13369344);                   // B*256 bf16
  __bf16* ws_fcW   = (__bf16*)(ws + 14417920);                   // 256*448 bf16
  __bf16* ws_Wih   = (__bf16*)(ws + 14647296);                   // 2*1024*256 bf16
  __bf16* ws_Whh   = (__bf16*)(ws + 15695872);                   // 2*1024*256 bf16

  // Weight conversions (once per call; amortized across all M-tiles)
  k_tobf<<<(256 * IN_ / 2 + 255) / 256, 256, 0, stream>>>(fcW, (unsigned*)ws_fcW, 256 * IN_ / 2);
  k_tobf<<<(L_ * 4 * H_ * H_ / 2 + 255) / 256, 256, 0, stream>>>(Wih, (unsigned*)ws_Wih, L_ * 4 * H_ * H_ / 2);
  k_tobf<<<(L_ * 4 * H_ * H_ / 2 + 255) / 256, 256, 0, stream>>>(Whh, (unsigned*)ws_Whh, L_ * 4 * H_ * H_ / 2);

  k_prep<<<B_, 64, 0, stream>>>(loc, tim, des, inter, prefer, remb, temb,
                                Wq, bq, Wk, bk, Wv, bv, Wo, bo, ln_g, ln_b, ws_in_bf);

  k_fcin<<<dim3(H_ / 64, B_ / 16), 32, 0, stream>>>(ws_in_bf, ws_fcW, fcb, ws_x_bf);

  for (int l = 0; l < L_; ++l) {
    k_tobf<<<(B_ * H_ / 2 + 255) / 256, 256, 0, stream>>>(
        cur_h + (size_t)l * B_ * H_, (unsigned*)ws_h_bf, B_ * H_ / 2);
    k_gates<<<dim3(4 * H_ / 64, B_ / 16), 32, 0, stream>>>(
        ws_x_bf, ws_h_bf,
        ws_Wih + (size_t)l * 4 * H_ * H_, ws_Whh + (size_t)l * 4 * H_ * H_,
        bih + (size_t)l * 4 * H_, bhh + (size_t)l * 4 * H_, ws_g);
    k_cell<<<(B_ * H_ + 255) / 256, 256, 0, stream>>>(
        ws_g, cur_c + (size_t)l * B_ * H_,
        o_nh + (size_t)l * B_ * H_, o_nc + (size_t)l * B_ * H_, ws_x_bf, ws_xf);
  }

  const size_t shmem = (size_t)((S_ + 1) * H_ + 3 * H_) * sizeof(float);  // ~209 KB < 320 KB/WGP
  (void)hipFuncSetAttribute((const void*)k_hist,
                            hipFuncAttributeMaxDynamicSharedMemorySize, (int)shmem);
  k_hist<<<B_, H_, shmem, stream>>>(hist, ws_xf, o_hist, o_ms);

  k_cand<<<B_, 256, 0, stream>>>(o_ms, cmask, outW, outb, o_cand);
  k_info<<<B_, E_, 0, stream>>>(o_ms, infW, infb, o_info);
}
